// Spike_LSTM_With_T_37632503447569
// MI455X (gfx1250) — matrix operand order repaired and checked by reading, not compiled, not run on hardware
//
#include <hip/hip_runtime.h>

// ---------------------------------------------------------------------------
// Spiking LSTM (2 layers) for MI455X / gfx1250.  L=256, TB=128, C=H=256.
//  * bf16 WMMA everywhere (h is provably binary {0,1} -> exact in bf16).
//  * Input projections for all L batched: G = X @ W_ih^T + b  (one big GEMM).
//  * Recurrence: ONE persistent kernel per layer (32 WGs). Wave owns a
//    16x16 (tb,h) patch across all 4 gate blocks, so the spike/cell update
//    is done in registers; c stays in VGPRs for all 256 steps. W_hh tiles
//    live in LDS (loaded once via async-to-LDS). One atomic grid barrier
//    per step.
// ---------------------------------------------------------------------------

typedef __attribute__((ext_vector_type(8)))  float   v8f;
typedef __attribute__((ext_vector_type(8)))  __bf16  v8bf;
typedef __attribute__((ext_vector_type(16))) __bf16  v16bf;
typedef __attribute__((ext_vector_type(4)))  int     v4i;

typedef __attribute__((address_space(1))) v4i* gv4i_ptr;
typedef __attribute__((address_space(3))) v4i* lv4i_ptr;

static constexpr int kL   = 256;
static constexpr int kTB  = 128;   // T*B
static constexpr int kC   = 256;   // input features (== H)
static constexpr int kH   = 256;
static constexpr int kG4  = 1024;  // 4*H
static constexpr int kM1  = kL * kTB;  // 32768
static constexpr int kTBH = kTB * kH;  // 32768
static constexpr int kNWG = 32;        // persistent grid size

__device__ __forceinline__ v16bf cat8(v8bf lo, v8bf hi) {
  return __builtin_shufflevector(lo, hi, 0, 1, 2, 3, 4, 5, 6, 7,
                                 8, 9, 10, 11, 12, 13, 14, 15);
}

// A fragment (16x32 bf16): lanes 0-15 row M=lane K={0..15}; lanes
// 16-31 row M=lane-16 K={16..31}. Two contiguous 16B loads.
__device__ __forceinline__ v16bf load_a_frag(const __bf16* __restrict__ aRow,
                                             int k0, int kA) {
  v8bf lo = *(const v8bf*)(aRow + k0 + kA);
  v8bf hi = *(const v8bf*)(aRow + k0 + kA + 8);
  return cat8(lo, hi);
}

// B fragment (32x16 bf16) from weights stored [N,K] row-major: lane n needs
// 16 contiguous K values at k0 (lanes 0-15) / k0+16 (lanes 16-31).
__device__ __forceinline__ v16bf load_b_frag_ptr(const __bf16* p) {
  v8bf lo = *(const v8bf*)(p);
  v8bf hi = *(const v8bf*)(p + 8);
  return cat8(lo, hi);
}

// 16-byte global -> LDS copy; async-tensor path when toolchain exposes it.
__device__ __forceinline__ void copy16_g2l(__bf16* dst_lds,
                                           const __bf16* src_glb) {
#if __has_builtin(__builtin_amdgcn_global_load_async_to_lds_b128)
  __builtin_amdgcn_global_load_async_to_lds_b128(
      (gv4i_ptr)(unsigned long long)src_glb,   // drop const, generic->global
      (lv4i_ptr)dst_lds,
      /*offset=*/0, /*cpol=*/0);
#else
  *(v8bf*)dst_lds = *(const v8bf*)src_glb;   // ds_store_b128 fallback
#endif
}

__device__ __forceinline__ void wait_async_copies() {
#if __has_builtin(__builtin_amdgcn_s_wait_asynccnt)
  __builtin_amdgcn_s_wait_asynccnt(0);
#else
  asm volatile("s_wait_asynccnt 0" ::: "memory");
#endif
}

// Device-scope sense-free grid barrier (monotone counter).
__device__ __forceinline__ void grid_barrier(unsigned* cnt, unsigned target) {
  __threadfence();          // make this wave's stores device-visible
  __syncthreads();
  if (threadIdx.x == 0) {
    __hip_atomic_fetch_add(cnt, 1u, __ATOMIC_RELEASE, __HIP_MEMORY_SCOPE_AGENT);
    while (__hip_atomic_load(cnt, __ATOMIC_ACQUIRE,
                             __HIP_MEMORY_SCOPE_AGENT) < target) {
      __builtin_amdgcn_s_sleep(1);
    }
  }
  __syncthreads();
}

// ---------------------------------------------------------------------------
// Batched input projection: G[M,1024] = A[M,256] @ W^T + bias
// grid = (M/16, 4), block = 128 (4 waves); wave owns a 16x64 tile.
// ---------------------------------------------------------------------------
__global__ __launch_bounds__(128)
void gemm_in_proj_kernel(const __bf16* __restrict__ A,
                         const __bf16* __restrict__ W,     // [1024,256] bf16
                         const float*  __restrict__ bias,  // [1024]
                         float* __restrict__ Gout) {
  const int lane  = threadIdx.x & 31;
  const int wave  = threadIdx.x >> 5;
  const int l15   = lane & 15;
  const int hi16  = (lane >> 4) & 1;
  const int tileM = blockIdx.x * 16;
  const int tileN = blockIdx.y * 256 + wave * 64;
  const int kA    = hi16 ? 16 : 0;
  const int kB    = hi16 ? 16 : 0;

  const __bf16* aRow = A + (size_t)(tileM + l15) * kC;
  __builtin_prefetch(aRow, 0, 0);   // global_prefetch_b8

  v8f acc[4];
#pragma unroll
  for (int s = 0; s < 4; ++s) {
    const float bv = bias[tileN + s * 16 + l15];
#pragma unroll
    for (int r = 0; r < 8; ++r) acc[s][r] = bv;
  }

#pragma unroll
  for (int k0 = 0; k0 < kC; k0 += 32) {
    const v16bf af = load_a_frag(aRow, k0, kA);
#pragma unroll
    for (int s = 0; s < 4; ++s) {
      const v16bf bf =
          load_b_frag_ptr(W + (size_t)(tileN + s * 16 + l15) * kC + k0 + kB);
      acc[s] = __builtin_amdgcn_wmma_f32_16x16x32_bf16(
          false, af, false, bf, (short)0, acc[s], false, false);
    }
  }

#pragma unroll
  for (int s = 0; s < 4; ++s) {
    const int n = tileN + s * 16 + l15;
#pragma unroll
    for (int r = 0; r < 8; ++r)
      Gout[(size_t)(tileM + hi16 * 8 + r) * kG4 + n] = acc[s][r];
  }
}

// ---------------------------------------------------------------------------
// Persistent recurrence kernel: whole L=256 scan for one layer.
// grid = 32 WGs x 128 threads (4 waves). WG w: h-tile = w&15 (16 cols),
// M-group = w>>4; wave v owns M-tile = (w>>4)*4+v. Wave's 4 accumulators
// are gates i,f,g,o for its 16x16 (tb,h) patch.
// ---------------------------------------------------------------------------
__global__ __launch_bounds__(128)
void lstm_scan_kernel(const float* __restrict__ G,     // [L*TB,1024] x@W+b
                      __bf16* __restrict__ Hbf,        // [(L+1)*TB*H] bf16
                      const __bf16* __restrict__ Whh,  // [1024,256] bf16
                      float* outF,                     // [L*TB*H] or null
                      float* __restrict__ stateH,      // [TB*H]
                      float* __restrict__ stateC,      // [TB*H]
                      unsigned* __restrict__ cnt) {
  __shared__ __bf16 sW[4 * 16 * kC];   // 32 KB: [gate*16+col][k]

  const int tid   = threadIdx.x;
  const int lane  = tid & 31;
  const int wave  = tid >> 5;
  const int l15   = lane & 15;
  const int hi16  = (lane >> 4) & 1;
  const int h0    = (blockIdx.x & 15) * 16;                 // h columns
  const int tileM = ((blockIdx.x >> 4) * 4 + wave) * 16;    // tb rows
  const int kA    = hi16 ? 16 : 0;
  const int kB    = hi16 ? 16 : 0;

  // --- Stage the 64 W_hh rows this WG needs into LDS (once, async). ---
  for (int i = tid; i < 64 * 32; i += 128) {       // 64 rows x 32 16B chunks
    const int nl = i >> 5, ch = i & 31;
    const int gate = nl >> 4, col = nl & 15;
    copy16_g2l(&sW[nl * kC + ch * 8],
               Whh + (size_t)(gate * kH + h0 + col) * kC + ch * 8);
  }
  wait_async_copies();
  __syncthreads();

  float creg[8];
#pragma unroll
  for (int r = 0; r < 8; ++r) creg[r] = 0.0f;

  const int rowBase = tileM + hi16 * 8;
  const int col     = h0 + l15;

  for (int l = 0; l < kL; ++l) {
    const float*  Gl   = G + (size_t)l * kTB * kG4;
    const __bf16* aRow = Hbf + (size_t)l * kTBH + (size_t)(tileM + l15) * kH;
    __builtin_prefetch(aRow, 0, 0);

    v8f acc[4];
#pragma unroll
    for (int s = 0; s < 4; ++s) {
      const int n = s * kH + col;            // gate s, column col
#pragma unroll
      for (int r = 0; r < 8; ++r)
        acc[s][r] = Gl[(size_t)(rowBase + r) * kG4 + n];
    }

#pragma unroll
    for (int k0 = 0; k0 < kH; k0 += 32) {
      const v16bf af = load_a_frag(aRow, k0, kA);
#pragma unroll
      for (int s = 0; s < 4; ++s) {
        const v16bf bf = load_b_frag_ptr(&sW[(s * 16 + l15) * kC + k0 + kB]);
        acc[s] = __builtin_amdgcn_wmma_f32_16x16x32_bf16(
            false, af, false, bf, (short)0, acc[s], false, false);
      }
    }

    // --- spike / cell update entirely in registers ---
    __bf16* Hn = Hbf + (size_t)(l + 1) * kTBH;
#pragma unroll
    for (int r = 0; r < 8; ++r) {
      const float iv = (acc[0][r] >= 0.0f) ? 1.0f : 0.0f;
      const float fv = (acc[1][r] >= 0.0f) ? 1.0f : 0.0f;
      const float gv = (acc[2][r] >= 0.0f) ? 1.0f : 0.0f;
      const float ov = (acc[3][r] >= 0.0f) ? 1.0f : 0.0f;
      float c = fv * creg[r] + iv * gv;
      c = fminf(c, 1.0f);
      creg[r] = c;
      const float h = ov * c;
      const size_t idx = (size_t)(rowBase + r) * kH + col;
      Hn[idx] = (__bf16)h;                          // exact: h in {0,1}
      if (outF) outF[(size_t)l * kTBH + idx] = h;
      if (l == kL - 1) { stateH[idx] = h; stateC[idx] = c; }
    }

    if (l != kL - 1)
      grid_barrier(cnt, (unsigned)(l + 1) * kNWG);
  }
}

__global__ __launch_bounds__(256)
void cvt_f32_bf16_kernel(const float* __restrict__ src,
                         __bf16* __restrict__ dst, int n) {
  int i = blockIdx.x * blockDim.x + threadIdx.x;
  const int stride = gridDim.x * blockDim.x;
  for (; i < n; i += stride) dst[i] = (__bf16)src[i];
}

// ---------------------------------------------------------------------------
extern "C" void kernel_launch(void* const* d_in, const int* in_sizes, int n_in,
                              void* d_out, int out_size, void* d_ws,
                              size_t ws_size, hipStream_t stream) {
  (void)in_sizes; (void)n_in; (void)out_size; (void)ws_size;
  const float* x     = (const float*)d_in[0];
  const float* W_ih0 = (const float*)d_in[1];
  const float* W_hh0 = (const float*)d_in[2];
  const float* b0    = (const float*)d_in[3];
  const float* W_ih1 = (const float*)d_in[4];
  const float* W_hh1 = (const float*)d_in[5];
  const float* b1    = (const float*)d_in[6];
  float* out = (float*)d_out;

  // Workspace carve-up (~170 MB):
  char* ws = (char*)d_ws;
  float* G = (float*)ws;                                    // [32768,1024] f32
  size_t off = (size_t)kM1 * kG4 * sizeof(float);
  __bf16* Xbf = (__bf16*)(ws + off);  off += (size_t)kM1 * kC * 2;
  __bf16* Hbf = (__bf16*)(ws + off);  off += (size_t)(kL + 1) * kTBH * 2;
  __bf16* Wih0b = (__bf16*)(ws + off); off += (size_t)kG4 * kC * 2;
  __bf16* Whh0b = (__bf16*)(ws + off); off += (size_t)kG4 * kC * 2;
  __bf16* Wih1b = (__bf16*)(ws + off); off += (size_t)kG4 * kC * 2;
  __bf16* Whh1b = (__bf16*)(ws + off); off += (size_t)kG4 * kC * 2;
  unsigned* cnt0 = (unsigned*)(ws + off); off += 128;
  unsigned* cnt1 = (unsigned*)(ws + off); off += 128;

  // Output layout: out1 [L,T,B,H], state_h [2,T,B,H], state_c [2,T,B,H]
  float* out1 = out;
  float* sh   = out + (size_t)kL * kTBH;
  float* sc   = sh + 2 * (size_t)kTBH;

  cvt_f32_bf16_kernel<<<1024, 256, 0, stream>>>(x, Xbf, kM1 * kC);
  cvt_f32_bf16_kernel<<<256, 256, 0, stream>>>(W_ih0, Wih0b, kG4 * kC);
  cvt_f32_bf16_kernel<<<256, 256, 0, stream>>>(W_hh0, Whh0b, kG4 * kC);
  cvt_f32_bf16_kernel<<<256, 256, 0, stream>>>(W_ih1, Wih1b, kG4 * kC);
  cvt_f32_bf16_kernel<<<256, 256, 0, stream>>>(W_hh1, Whh1b, kG4 * kC);
  (void)hipMemsetAsync(Hbf, 0, (size_t)kTBH * 2, stream);   // h_{-1} = 0
  (void)hipMemsetAsync(cnt0, 0, 256, stream);               // barrier counters

  const dim3 gridBig(kM1 / 16, 4);

  // ---------------- Layer 0 ----------------
  gemm_in_proj_kernel<<<gridBig, 128, 0, stream>>>(Xbf, Wih0b, b0, G);
  lstm_scan_kernel<<<kNWG, 128, 0, stream>>>(G, Hbf, Whh0b,
                                             /*outF=*/nullptr, sh, sc, cnt0);

  // ---------------- Layer 1 ----------------
  // Layer-0 h history (Hbf slots 1..L) is the layer-1 input; the batched
  // GEMM consumes it fully before the scan kernel recycles Hbf.
  gemm_in_proj_kernel<<<gridBig, 128, 0, stream>>>(Hbf + kTBH, Wih1b, b1, G);
  lstm_scan_kernel<<<kNWG, 128, 0, stream>>>(G, Hbf, Whh1b,
                                             out1, sh + kTBH, sc + kTBH, cnt1);
}
